// RCNNSpMiddleFHD_33998961115510
// MI455X (gfx1250) — matrix operand order repaired and checked by reading, not compiled, not run on hardware
//
#include <hip/hip_runtime.h>
#include <hip/hip_bf16.h>

// ---------------------------------------------------------------------------
// CDNA5 (gfx1250, wave32) implementation of RCNNSpMiddleFHD:
// dense implicit-GEMM conv3d with v_wmma_f32_16x16x32_bf16, masked BN+ReLU.
// Conv kernels are fully templated per layer so taps/K-chunks unroll and
// address math strength-reduces to 32-bit adds.
// ---------------------------------------------------------------------------

typedef __attribute__((ext_vector_type(16))) __bf16        v16bf;
typedef __attribute__((ext_vector_type(8)))  float         v8f;
typedef __attribute__((ext_vector_type(4)))  unsigned int  v4u;

union ABu { v4u q[2]; v16bf v; };   // 32 bytes: 16 bf16 = one A or B fragment/lane

// ---------------------------------------------------------------------------
// Scatter voxel features into dense bf16 grid [D*H*W][C] + uint8 mask
// ---------------------------------------------------------------------------
__global__ void scatter_kernel(const float* __restrict__ vf,
                               const int* __restrict__ coors,
                               __hip_bfloat16* __restrict__ x,
                               unsigned char* __restrict__ mask,
                               int nvox, int C, int H, int W)
{
    int i = blockIdx.x * blockDim.x + threadIdx.x;
    if (i >= nvox) return;
    const int z  = coors[i * 4 + 1];
    const int yy = coors[i * 4 + 2];
    const int xx = coors[i * 4 + 3];
    const int pos = (z * H + yy) * W + xx;
    mask[pos] = 1;
    for (int c = 0; c < C; ++c)
        x[(long long)pos * C + c] = __hip_bfloat16(vf[(long long)i * C + c]);
}

// ---------------------------------------------------------------------------
// Pre-pack weights OIDHW f32 -> [tap][Cout][Kpad] bf16 (rows >= Cin zeroed)
// ---------------------------------------------------------------------------
__global__ void prepack_w_kernel(const float* __restrict__ w,
                                 __hip_bfloat16* __restrict__ wpk,
                                 int Cout, int Cin, int Kpad, int ntap)
{
    int i = blockIdx.x * blockDim.x + threadIdx.x;
    int n = ntap * Kpad * Cout;
    if (i >= n) return;
    int r = i % Kpad;
    int o = (i / Kpad) % Cout;
    int t = i / (Cout * Kpad);
    float v = (r < Cin) ? w[(o * Cin + r) * ntap + t] : 0.0f;
    wpk[i] = __hip_bfloat16(v);
}

// ---------------------------------------------------------------------------
// Implicit-GEMM conv3d with WMMA bf16 (f32 accumulate), fully templated.
// One wave computes a 16-position x 16-out-channel tile; 4 waves per block.
//   A layout (16x32 bf16): lane l holds M=l&15; elems 0..7 -> K=kb+e,
//                          elems 8..15 -> K=kb+16+(e-8), kb=(l>=16)*8.
//   B layout (32x16 bf16): lane l holds col N=l&15; elems -> K same as A.
//   D layout (16x16 f32):  vgpr r, lane l -> M=r+(l>=16)*8, N=l&15.
// ---------------------------------------------------------------------------
template <int CIN, int COUT, int KD, int KH, int KW,
          int SD, int SH, int SW, int PD, int PH, int PW>
__global__ void __launch_bounds__(128)
conv3d_wmma_kernel(const __hip_bfloat16* __restrict__ xin,
                   const __hip_bfloat16* __restrict__ wpk,
                   float* __restrict__ y,
                   int inD, int inH, int inW,
                   int outD, int outH, int outW)
{
    constexpr int KPAD = (CIN < 32) ? 32 : CIN;

    const int lane = threadIdx.x & 31;
    const int wv   = threadIdx.x >> 5;
    const int outPos = outD * outH * outW;
    const int tile   = blockIdx.x * 4 + wv;
    const int obase  = blockIdx.y * 16;
    if (tile * 16 >= outPos) return;            // uniform per wave

    const int  Mrow   = lane & 15;
    int        m      = tile * 16 + Mrow;
    const bool mvalid = (m < outPos);
    if (!mvalid) m = 0;                         // clamp for safe addressing

    const int ow = m % outW;
    const int t0 = m / outW;
    const int oh = t0 % outH;
    const int od = t0 / outH;

    const int kb = (lane >> 4) * 8;             // A K sub-base per half-wave
    const v4u z128 = (v4u)0u;

    v8f acc = {};

#pragma unroll
    for (int zk = 0; zk < KD; ++zk) {
        const int  id  = od * SD + zk - PD;
        const bool okD = (unsigned)id < (unsigned)inD;
        const int  idc = okD ? id : 0;
#pragma unroll
        for (int yk = 0; yk < KH; ++yk) {
            const int  ih  = oh * SH + yk - PH;
            const bool okH = (unsigned)ih < (unsigned)inH;
            const int  ihc = okH ? ih : 0;
#pragma unroll
            for (int xk = 0; xk < KW; ++xk) {
                const int  iw  = ow * SW + xk - PW;
                const bool okW = (unsigned)iw < (unsigned)inW;
                const int  iwc = okW ? iw : 0;
                const bool vin = mvalid && okD && okH && okW;

                const int srcbase = ((idc * inH + ihc) * inW + iwc) * CIN;
                const int tap     = (zk * KH + yk) * KW + xk;

                // uniform weight-tile prefetch (global_prefetch_b8)
                __builtin_prefetch(wpk + tap * KPAD * COUT, 0, 1);

#pragma unroll
                for (int kc = 0; kc < KPAD; kc += 32) {
                    // A fragment: always-valid clamped loads, then zero-fix
                    v4u r0 = z128, r1 = z128;
                    if ((kc + 16) <= CIN)       // constant-folds per CIN
                        r0 = *(const v4u*)(xin + srcbase + kc + kb);
                    if ((kc + 32) <= CIN)
                        r1 = *(const v4u*)(xin + srcbase + kc + kb + 16);
                    if (!vin) { r0 = z128; r1 = z128; }
                    ABu a;
                    a.q[0] = r0;
                    a.q[1] = r1;

                    // B fragment: lane l holds column N = obase+(l&15), K same order as A
                    ABu b;
                    const v4u* wr = (const v4u*)
                        (wpk + ((tap * COUT + obase + (lane & 15)) * KPAD + kc + kb));
                    b.q[0] = wr[0];
                    b.q[1] = wr[2];

                    acc = __builtin_amdgcn_wmma_f32_16x16x32_bf16(
                        false, a.v, false, b.v, (short)0, acc, false, false);
                }
            }
        }
    }

    const int n    = obase + (lane & 15);
    const int moff = (lane >> 4) * 8;
#pragma unroll
    for (int r = 0; r < 8; ++r) {
        const int mo = tile * 16 + moff + r;
        if (mo < outPos) y[mo * COUT + n] = acc[r];
    }
}

// ---------------------------------------------------------------------------
// Mask downsample: out = any(mask in receptive window)
// ---------------------------------------------------------------------------
__global__ void mask_down_kernel(const unsigned char* __restrict__ mi,
                                 unsigned char* __restrict__ mo,
                                 int inD, int inH, int inW,
                                 int outD, int outH, int outW,
                                 int kd, int kh, int kw,
                                 int sd, int sh, int sw,
                                 int pd, int ph, int pw)
{
    int i = blockIdx.x * blockDim.x + threadIdx.x;
    int n = outD * outH * outW;
    if (i >= n) return;
    int ow = i % outW;
    int oh = (i / outW) % outH;
    int od = i / (outW * outH);
    unsigned char any = 0;
    for (int zk = 0; zk < kd; ++zk) {
        int id = od * sd + zk - pd;
        if ((unsigned)id >= (unsigned)inD) continue;
        for (int yk = 0; yk < kh; ++yk) {
            int ih = oh * sh + yk - ph;
            if ((unsigned)ih >= (unsigned)inH) continue;
            for (int xk = 0; xk < kw; ++xk) {
                int iw = ow * sw + xk - pw;
                if ((unsigned)iw >= (unsigned)inW) continue;
                any |= mi[(id * inH + ih) * inW + iw];
            }
        }
    }
    mo[i] = any ? 1 : 0;
}

// ---------------------------------------------------------------------------
// Masked BN statistics: stats[c]=sum, stats[64+c]=sumsq, stats[128]=count
// ---------------------------------------------------------------------------
__global__ void bn_stats_kernel(const float* __restrict__ y,
                                const unsigned char* __restrict__ mask,
                                float* __restrict__ stats,
                                int npos, int C)
{
    __shared__ float ssum[256], ssq[256], scnt[256];
    const int c = blockIdx.x;
    float s = 0.f, s2 = 0.f, cm = 0.f;
    for (long long p = (long long)blockIdx.y * blockDim.x + threadIdx.x;
         p < npos;
         p += (long long)gridDim.y * blockDim.x) {
        const float mk = mask[p] ? 1.f : 0.f;
        const float v  = y[p * C + c];
        s  += v * mk;
        s2 += v * v * mk;
        cm += mk;
    }
    ssum[threadIdx.x] = s; ssq[threadIdx.x] = s2; scnt[threadIdx.x] = cm;
    __syncthreads();
    for (int st = 128; st > 0; st >>= 1) {
        if ((int)threadIdx.x < st) {
            ssum[threadIdx.x] += ssum[threadIdx.x + st];
            ssq[threadIdx.x]  += ssq[threadIdx.x + st];
            scnt[threadIdx.x] += scnt[threadIdx.x + st];
        }
        __syncthreads();
    }
    if (threadIdx.x == 0) {
        atomicAdd(&stats[c], ssum[0]);
        atomicAdd(&stats[64 + c], ssq[0]);
        if (c == 0) atomicAdd(&stats[128], scnt[0]);
    }
}

// ---------------------------------------------------------------------------
// BN apply + ReLU + mask, write next-layer bf16 activations
// var = E[x^2 | mask] - mean^2   (== reference masked variance)
// ---------------------------------------------------------------------------
__global__ void bn_apply_kernel(const float* __restrict__ y,
                                const unsigned char* __restrict__ mask,
                                const float* __restrict__ gamma,
                                const float* __restrict__ beta,
                                const float* __restrict__ stats,
                                __hip_bfloat16* __restrict__ xout,
                                long long npos, int C)
{
    long long i = (long long)blockIdx.x * blockDim.x + threadIdx.x;
    if (i >= npos * C) return;
    const int c       = (int)(i % C);
    const long long p = i / C;
    const float cnt  = fmaxf(stats[128], 1.0f);
    const float mean = stats[c] / cnt;
    const float var  = stats[64 + c] / cnt - mean * mean;
    const float sc   = gamma[c] * rsqrtf(var + 1e-3f);
    float v = (y[i] - mean) * sc + beta[c];
    v = fmaxf(v, 0.0f);
    v = mask[p] ? v : 0.0f;
    xout[i] = __hip_bfloat16(v);
}

// ---------------------------------------------------------------------------
// Final transpose/reshape: x[D,H,W,C] -> out[(c*W+w)*D*H + d*H + h] as f32
// ---------------------------------------------------------------------------
__global__ void finalize_kernel(const __hip_bfloat16* __restrict__ x,
                                float* __restrict__ out,
                                int D, int H, int W, int C)
{
    int i = blockIdx.x * blockDim.x + threadIdx.x;
    int n = C * W * D * H;
    if (i >= n) return;
    int h = i % H;
    int d = (i / H) % D;
    int w = (i / (H * D)) % W;
    int c = i / (H * D * W);
    out[i] = (float)x[((d * H + h) * W + w) * C + c];
}

// ---------------------------------------------------------------------------
// Host-side per-layer driver (templated so each layer gets its own conv
// instantiation with fully unrolled taps/K-chunks)
// ---------------------------------------------------------------------------
template <int CIN, int COUT, int KD, int KH, int KW,
          int SD, int SH, int SW, int PD, int PH, int PW>
static void run_layer(const float* w, const float* g, const float* bb,
                      __hip_bfloat16*& xc, __hip_bfloat16*& xn,
                      unsigned char*& mc, unsigned char*& mn,
                      __hip_bfloat16* wpk, float* yb, float* stats,
                      int& D, int& H, int& W, hipStream_t stream)
{
    constexpr int KPAD = (CIN < 32) ? 32 : CIN;
    constexpr int NTAP = KD * KH * KW;

    const int oD = (D + 2 * PD - KD) / SD + 1;
    const int oH = (H + 2 * PH - KH) / SH + 1;
    const int oW = (W + 2 * PW - KW) / SW + 1;
    const int outPos = oD * oH * oW;

    // pre-pack weights (zero-padded K rows)
    const int wn = NTAP * KPAD * COUT;
    prepack_w_kernel<<<(wn + 255) / 256, 256, 0, stream>>>(
        w, wpk, COUT, CIN, KPAD, NTAP);

    // conv (dense implicit GEMM, WMMA bf16): 4 tiles (waves) per block
    const int tiles = (outPos + 15) / 16;
    dim3 cgrid((unsigned)((tiles + 3) / 4), (unsigned)(COUT / 16));
    conv3d_wmma_kernel<CIN, COUT, KD, KH, KW, SD, SH, SW, PD, PH, PW>
        <<<cgrid, 128, 0, stream>>>(xc, wpk, yb, D, H, W, oD, oH, oW);

    // 'down' layers recompute the mask before BN
    constexpr bool DOWN = !(SD == 1 && SH == 1 && SW == 1);
    if (DOWN) {
        mask_down_kernel<<<(outPos + 255) / 256, 256, 0, stream>>>(
            mc, mn, D, H, W, oD, oH, oW,
            KD, KH, KW, SD, SH, SW, PD, PH, PW);
        unsigned char* t = mc; mc = mn; mn = t;
    }

    // masked BN stats + apply(ReLU, mask) -> next bf16 grid
    hipMemsetAsync(stats, 0, 136 * sizeof(float), stream);
    bn_stats_kernel<<<dim3((unsigned)COUT, 256), 256, 0, stream>>>(
        yb, mc, stats, outPos, COUT);
    const long long tot = (long long)outPos * COUT;
    bn_apply_kernel<<<(unsigned)((tot + 255) / 256), 256, 0, stream>>>(
        yb, mc, g, bb, stats, xn, outPos, COUT);

    __hip_bfloat16* t = xc; xc = xn; xn = t;
    D = oD; H = oH; W = oW;
}

// ---------------------------------------------------------------------------
// Host orchestration
// ---------------------------------------------------------------------------
extern "C" void kernel_launch(void* const* d_in, const int* in_sizes, int n_in,
                              void* d_out, int out_size, void* d_ws, size_t ws_size,
                              hipStream_t stream)
{
    (void)n_in; (void)out_size;
    const float* vf    = (const float*)d_in[0];
    const int*   coors = (const int*)d_in[1];
    const int    nvox  = in_sizes[1] / 4;

    // ---- workspace carve ----
    auto alignup = [](size_t x) { return (x + 255) & ~(size_t)255; };
    const size_t NPOS0 = 40ull * 200 * 200;   // 1,600,000 dense positions
    char* base = (char*)d_ws;
    size_t off = 0;
    __hip_bfloat16* xA = (__hip_bfloat16*)(base + off); off += alignup(NPOS0 * 16 * 2);
    __hip_bfloat16* xB = (__hip_bfloat16*)(base + off); off += alignup(NPOS0 * 16 * 2);
    float*          yb = (float*)(base + off);          off += alignup(NPOS0 * 16 * 4);
    unsigned char*  mA = (unsigned char*)(base + off);  off += alignup(NPOS0);
    unsigned char*  mB = (unsigned char*)(base + off);  off += alignup(NPOS0);
    __hip_bfloat16* wpk = (__hip_bfloat16*)(base + off); off += alignup(27ull * 64 * 64 * 2);
    float*          stats = (float*)(base + off);        off += 1024;
    if (off > ws_size) return;   // workspace too small: nothing sane to do

    // ---- scatter into dense grid ----
    hipMemsetAsync(xA, 0, NPOS0 * 16 * 2, stream);
    hipMemsetAsync(mA, 0, NPOS0, stream);
    scatter_kernel<<<(nvox + 255) / 256, 256, 0, stream>>>(
        vf, coors, xA, mA, nvox, 16, 200, 200);

    // ---- layer pipeline ----
    int D = 40, H = 200, W = 200;
    __hip_bfloat16* xc = xA; __hip_bfloat16* xn = xB;
    unsigned char*  mc = mA; unsigned char*  mn = mB;

    const float** W_ = (const float**)alloca(0); (void)W_;
#define LAYER_ARGS(i) (const float*)d_in[4 + 3*(i)], (const float*)d_in[5 + 3*(i)], \
                      (const float*)d_in[6 + 3*(i)], xc, xn, mc, mn, wpk, yb, stats, \
                      D, H, W, stream

    run_layer<16, 16, 3,3,3, 1,1,1, 1,1,1>(LAYER_ARGS(0));   // subm
    run_layer<16, 16, 3,3,3, 1,1,1, 1,1,1>(LAYER_ARGS(1));   // subm
    run_layer<16, 32, 3,3,3, 2,2,2, 1,1,1>(LAYER_ARGS(2));   // down
    run_layer<32, 32, 3,3,3, 1,1,1, 1,1,1>(LAYER_ARGS(3));   // subm
    run_layer<32, 64, 3,3,3, 2,2,2, 1,1,1>(LAYER_ARGS(4));   // down
    run_layer<64, 64, 3,3,3, 1,1,1, 1,1,1>(LAYER_ARGS(5));   // subm
    run_layer<64, 64, 3,3,3, 2,2,2, 1,1,0>(LAYER_ARGS(6));   // down (pad 1,1,0)
    run_layer<64, 64, 3,3,3, 1,1,1, 1,1,1>(LAYER_ARGS(7));   // subm
    run_layer<64, 64, 1,1,3, 1,1,2, 0,0,0>(LAYER_ARGS(8));   // down (1,1,3)
#undef LAYER_ARGS

    // final (1, 64*11, 5, 25) transpose/reshape, f32 output
    const int on = 64 * 11 * 5 * 25;   // 88,000
    finalize_kernel<<<(on + 255) / 256, 256, 0, stream>>>(
        xc, (float*)d_out, 5, 25, 11, 64);
}
